// SACRSN_v39_23536420782589
// MI455X (gfx1250) — matrix operand order repaired and checked by reading, not compiled, not run on hardware
//
#include <hip/hip_runtime.h>
#include <hip/hip_bf16.h>

// ---------- problem constants ----------
#define BB      64
#define TT      32
#define DD      64
#define D2      128
#define VV      32000
#define SLOTS   64
#define MAXREC  6
#define BT      (BB * TT)          // 2048
#define LOGITS_N ((size_t)BT * VV) // 65,536,000

typedef __attribute__((ext_vector_type(16))) __bf16 v16bf;
typedef __attribute__((ext_vector_type(8)))  __bf16 v8bf;
typedef __attribute__((ext_vector_type(8)))  float  v8f;

#if defined(__has_builtin)
# if __has_builtin(__builtin_amdgcn_ds_swizzle)
#  define HAVE_SWZ 1
# endif
#endif
#ifndef HAVE_SWZ
# define HAVE_SWZ 0
#endif

// ---------- async Global->LDS copy (CDNA5, tracked by ASYNCcnt) ----------
// Per-lane: LDS[lds_addr] = MEM[gaddr] (16 bytes).  ISA §10.7 / §15.18 opcode 98.
__device__ inline void async_b128_to_lds(unsigned lds_addr, const void* gptr) {
  asm volatile("global_load_async_to_lds_b128 %0, %1, off"
               :: "v"(lds_addr), "v"((unsigned long long)(uintptr_t)gptr)
               : "memory");
}
__device__ inline void async_wait0() {
  asm volatile("s_wait_asynccnt 0x0" ::: "memory");
}
// stage a 16KB tile (64 rows x 256B) into LDS with a 128-thread block
__device__ inline void stage_tile_16k(const __bf16* g, unsigned lds_base, int tid) {
#pragma unroll
  for (int it = 0; it < 8; ++it) {
    int idx = it * 128 + tid;                 // 1024 chunks of 16B
    async_b128_to_lds(lds_base + idx * 16, g + idx * 8);
  }
  async_wait0();                              // this wave's asyncs done
  __syncthreads();                            // all waves' asyncs visible
}

// ---------- WMMA fragment helpers (CDNA5 16x16x32 bf16 layouts) ----------
__device__ inline v16bf ld16(const __bf16* p0, const __bf16* p1) {
  v8bf a = *(const v8bf*)p0;   // 16B aligned chunks
  v8bf b = *(const v8bf*)p1;
  return __builtin_shufflevector(a, b, 0,1,2,3,4,5,6,7,8,9,10,11,12,13,14,15);
}

// A matrix 16x32 bf16: lane m=lane&15; e -> k0+half*16+e
__device__ inline v16bf frag_a(const __bf16* A, int lda, int m0, int k0) {
  int lane = threadIdx.x & 31;
  int half = lane >> 4;
  int m = m0 + (lane & 15);
  const __bf16* base = A + (size_t)m * lda + k0 + half * 16;
  return ld16(base, base + 8);
}

// B matrix 32x16 bf16 from row-major W[n][k]: lane n=n0+(lane&15); e -> k0+half*16+e
__device__ inline v16bf frag_b(const __bf16* W, int ldw, int n0, int k0) {
  int lane = threadIdx.x & 31;
  int half = lane >> 4;
  int n = n0 + (lane & 15);
  const __bf16* base = W + (size_t)n * ldw + k0 + half * 16;
  return ld16(base, base + 8);
}

__device__ inline unsigned int fkey(float f) {   // order-preserving float->uint
  unsigned int u = __float_as_uint(f);
  return (u & 0x80000000u) ? ~u : (u | 0x80000000u);
}

__device__ inline unsigned long long shfl_xor_u64(unsigned long long v, int mask) {
  unsigned int lo = (unsigned int)v, hi = (unsigned int)(v >> 32);
  lo = __shfl_xor(lo, mask, 32);
  hi = __shfl_xor(hi, mask, 32);
  return ((unsigned long long)hi << 32) | lo;
}

#if HAVE_SWZ
#define SWZ_STAGE(key, MASK) do {                                                          \
    unsigned int _lo = (unsigned int)__builtin_amdgcn_ds_swizzle(                          \
        (int)(unsigned int)(key), ((MASK) << 10) | 31);                                    \
    unsigned int _hi = (unsigned int)__builtin_amdgcn_ds_swizzle(                          \
        (int)(unsigned int)((key) >> 32), ((MASK) << 10) | 31);                            \
    unsigned long long _o = ((unsigned long long)_hi << 32) | _lo;                         \
    (key) = (_o < (key)) ? _o : (key);                                                     \
  } while (0)
#else
#define SWZ_STAGE(key, MASK) do {                                                          \
    unsigned long long _o = shfl_xor_u64((key), (MASK));                                   \
    (key) = (_o < (key)) ? _o : (key);                                                     \
  } while (0)
#endif

// ---------- one-time prep ----------
__global__ void k_cvt_bf16(const float* a, __bf16* o, int n) {
  int i = blockIdx.x * 256 + threadIdx.x;
  if (i < n) o[i] = (__bf16)a[i];
}

__global__ void k_cvt_neg_bf16(const float* a, __bf16* o, int n) {
  int i = blockIdx.x * 256 + threadIdx.x;
  if (i < n) o[i] = (__bf16)(-a[i]);
}

__global__ void k_c2(const float* cb, float* c2) {
  int v = blockIdx.x * 256 + threadIdx.x;
  if (v < VV) {
    const float* p = cb + (size_t)v * D2;
    float s = 0.f;
    for (int c = 0; c < D2; ++c) s += p[c] * p[c];
    c2[v] = s;
  }
}

__global__ void k_zero(float* p, int n) {
  int i = blockIdx.x * 256 + threadIdx.x;
  if (i < n) p[i] = 0.f;
}

// ---------- outer step: g += embedding(x_t), reset inner-scan carries ----------
__global__ void k_outer(int t, const int* xseq, const float* emb,
                        float* gr, float* gi, float* cum, float* pen,
                        float* scal, unsigned int* activep) {
  int i = blockIdx.x * 256 + threadIdx.x;
  if (i < BB * DD) {
    int b = i >> 6, d = i & 63;
    int v = xseq[b * TT + t];
    gr[i] += emb[(size_t)v * D2 + d];
    gi[i] += emb[(size_t)v * D2 + DD + d];
  }
  if (blockIdx.x == 0) {
    if (threadIdx.x < BB) { cum[threadIdx.x] = 0.f; pen[threadIdx.x] = 0.f; }
    if (threadIdx.x == 0) { *activep = 1u; scal[3] = 0.f; scal[4] = 0.f; }
  }
}

// ---------- inner step: LayerNorm + zf (f32/bf16) + n (f32/bf16) + |zf|^2 + argmin init --
__global__ void k_pre(const float* gr, const float* gi,
                      const float* nrw, const float* nrb,
                      const float* niw, const float* nib,
                      float* nr, float* ni, __bf16* nr_bf, __bf16* ni_bf,
                      float* zf, __bf16* zfb, float* zf2,
                      unsigned long long* keys) {
  int tid = threadIdx.x;
  if (tid < 128) {
    int a = tid >> 6, b = tid & 63;
    const float* x = (a ? gi : gr) + b * DD;
    const float* w = a ? niw : nrw;
    const float* bi = a ? nib : nrb;
    float m = 0.f;
    for (int d = 0; d < DD; ++d) m += x[d];
    m *= (1.f / DD);
    float v = 0.f;
    for (int d = 0; d < DD; ++d) { float t = x[d] - m; v += t * t; }
    v *= (1.f / DD);
    float inv = rsqrtf(v + 1e-5f);
    float* o = (a ? ni : nr) + b * DD;
    __bf16* ob = (a ? ni_bf : nr_bf) + b * DD;
    for (int d = 0; d < DD; ++d) {
      float y = (x[d] - m) * inv * w[d] + bi[d];
      o[d] = y;
      ob[d] = (__bf16)y;
      zf[b * D2 + a * DD + d] = y;
      zfb[b * D2 + a * DD + d] = (__bf16)y;
    }
  }
  __syncthreads();
  if (tid < BB) {
    const float* z = zf + tid * D2;
    float s = 0.f;
    for (int c = 0; c < D2; ++c) s += z[c] * z[c];
    zf2[tid] = s;
    keys[tid] = ~0ull;
  }
}

// ---------- inner step: complex qkv projections on the matrix pipe ----------
__global__ void k_qkv(const __bf16* nr_bf, const __bf16* ni_bf,
                      const __bf16* wrb, const __bf16* wib, const __bf16* nwib,
                      const float* qkv_br, const float* qkv_bi,
                      float* qkvbuf) {
  int tid = threadIdx.x;
  int wave = tid >> 5, lane = tid & 31, half = lane >> 4, l15 = lane & 15;
  for (int i = 0; i < 12; ++i) {
    int tt = wave * 12 + i;              // 96 tiles over 8 waves
    int mat = tt >> 4;                   // j*2+im
    int within = tt & 15;
    int m0 = (within >> 2) * 16;         // batch-row tile
    int n0 = (within & 3) * 16;          // output-dim tile
    int j = mat >> 1, im = mat & 1;
    const __bf16* X1 = im ? ni_bf : nr_bf;
    const __bf16* X2 = im ? nr_bf : ni_bf;
    const __bf16* W1 = wrb + (size_t)j * DD * DD;
    const __bf16* W2 = (im ? wib : nwib) + (size_t)j * DD * DD;
    v8f acc = {};
#pragma unroll
    for (int k0 = 0; k0 < DD; k0 += 32)
      acc = __builtin_amdgcn_wmma_f32_16x16x32_bf16(false, frag_a(X1, DD, m0, k0),
                                                    false, frag_b(W1, DD, n0, k0),
                                                    (short)0, acc, false, false);
#pragma unroll
    for (int k0 = 0; k0 < DD; k0 += 32)
      acc = __builtin_amdgcn_wmma_f32_16x16x32_bf16(false, frag_a(X2, DD, m0, k0),
                                                    false, frag_b(W2, DD, n0, k0),
                                                    (short)0, acc, false, false);
    int n = n0 + l15;
    float br = qkv_br[j * DD + n], bi = qkv_bi[j * DD + n];
    float bias = im ? (br + bi) : (br - bi);
#pragma unroll
    for (int r = 0; r < 8; ++r) {
      int m = m0 + r + 8 * half;
      qkvbuf[((size_t)mat * BB + m) * DD + n] = acc[r] + bias;
    }
  }
}

// ---------- inner step: Hermitian gate + memory read (per-b block) ----------
__global__ void k_mid(const float* nr, const float* ni, const float* qkvbuf,
                      const float* mem_r, const float* mem_i,
                      float* attn_r, float* attn_i, float* read_r, float* read_i) {
  __shared__ float nrs[DD], nis[DD];
  __shared__ float qkv[6][DD];      // qr qi kr ki vr vi
  __shared__ float sim[SLOTS];
  __shared__ float red, ag;
  int b = blockIdx.x, tid = threadIdx.x;
  if (tid < DD) { nrs[tid] = nr[b * DD + tid]; nis[tid] = ni[b * DD + tid]; }
  if (tid == 0) red = 0.f;
  for (int o = tid; o < 6 * DD; o += 256)
    qkv[o >> 6][o & 63] = qkvbuf[((size_t)(o >> 6) * BB + b) * DD + (o & 63)];
  __syncthreads();
  if (tid < DD) atomicAdd(&red, qkv[0][tid] * qkv[2][tid] + qkv[1][tid] * qkv[3][tid]);
  __syncthreads();
  if (tid == 0) ag = 1.f / (1.f + expf(-red));
  __syncthreads();
  if (tid < DD) {
    attn_r[b * DD + tid] = qkv[4][tid] * ag;
    attn_i[b * DD + tid] = qkv[5][tid] * ag;
    const float* mr = mem_r + ((size_t)b * SLOTS + tid) * DD;
    const float* mi = mem_i + ((size_t)b * SLOTS + tid) * DD;
    float acc = 0.f;
    for (int d = 0; d < DD; ++d) acc += mr[d] * nrs[d] + mi[d] * nis[d];
    sim[tid] = acc;
  }
  __syncthreads();
  if (tid == 0) {
    float mx = sim[0];
    for (int s = 1; s < SLOTS; ++s) mx = fmaxf(mx, sim[s]);
    float sum = 0.f;
    for (int s = 0; s < SLOTS; ++s) { sim[s] = expf(sim[s] - mx); sum += sim[s]; }
    float inv = 1.f / sum;
    for (int s = 0; s < SLOTS; ++s) sim[s] *= inv;
  }
  __syncthreads();
  if (tid < DD) {
    float ar = 0.f, ai = 0.f;
    for (int s = 0; s < SLOTS; ++s) {
      ar += sim[s] * mem_r[((size_t)b * SLOTS + s) * DD + tid];
      ai += sim[s] * mem_i[((size_t)b * SLOTS + s) * DD + tid];
    }
    read_r[b * DD + tid] = ar;
    read_i[b * DD + tid] = ai;
  }
}

// ---------- inner step: VQ distance GEMM (WMMA bf16) + fused argmin ----------
// zf A-tile (16KB) staged once per block into LDS via async Global->LDS DMA.
__global__ void k_vq(const __bf16* zfb, const float* zf2,
                     const __bf16* cb, const float* c2,
                     unsigned long long* keys) {
  __shared__ __attribute__((aligned(16))) __bf16 Azf[BB * D2];   // 16KB
  __shared__ unsigned long long smin[BB];
  int tid = threadIdx.x;
  if (tid < BB) smin[tid] = ~0ull;
  stage_tile_16k(zfb, (unsigned)(uintptr_t)&Azf[0], tid);        // includes barrier
  const __bf16* Al = (const __bf16*)Azf;
  int wave = tid >> 5, lane = tid & 31, half = lane >> 4, l15 = lane & 15;
  int n0 = blockIdx.x * 64 + wave * 16;
  int n  = n0 + l15;
  float cn = c2[n];
#pragma unroll
  for (int mt = 0; mt < 4; ++mt) {
    int m0 = mt * 16;
    v8f acc = {};
#pragma unroll
    for (int kk = 0; kk < 4; ++kk) {
      v16bf a = frag_a(Al, D2, m0, kk * 32);     // ds_load from staged tile
      v16bf b = frag_b(cb, D2, n0, kk * 32);
      acc = __builtin_amdgcn_wmma_f32_16x16x32_bf16(false, a, false, b,
                                                    (short)0, acc, false, false);
    }
#pragma unroll
    for (int r = 0; r < 8; ++r) {
      int m = m0 + r + 8 * half;                 // C layout: VGPR r -> M=r (+8 hi half)
      float dist = zf2[m] + cn - 2.0f * acc[r];
      unsigned long long key =
          ((unsigned long long)fkey(dist) << 32) | (unsigned int)n;
      SWZ_STAGE(key, 1);
      SWZ_STAGE(key, 2);
      SWZ_STAGE(key, 4);
      SWZ_STAGE(key, 8);
      if (l15 == 0) atomicMin(&smin[m], key);    // LDS pre-reduce (ds_min_u64)
    }
  }
  __syncthreads();
  if (tid < BB) atomicMin(&keys[tid], smin[tid]); // global argmin per batch row
}

// ---------- inner step: losses, gates, state update, halting ----------
__global__ void k_post(int r, const float* zf, const float* cb_f32,
                       const unsigned long long* keys,
                       const float* gate_w, const float* gate_b,
                       const float* attn_r, const float* attn_i,
                       const float* read_r, const float* read_i,
                       const float* nr, const float* ni,
                       float* gr, float* gi, float* cum, float* pen,
                       float* scal, unsigned int* activep) {
  __shared__ int idx[BB];
  __shared__ float gl[BB][3];
  __shared__ float vqs, ent, cmean, pensum;
  __shared__ unsigned int act;
  int tid = threadIdx.x;
  if (tid == 0) { vqs = 0.f; ent = 0.f; cmean = 0.f; pensum = 0.f; act = *activep; }
  __syncthreads();
  if (tid < BB) idx[tid] = (int)(keys[tid] & 0xffffffffu);
  __syncthreads();
  float p = 0.f;
  for (int o = tid; o < BB * D2; o += 256) {
    int b = o >> 7, c = o & 127;
    float d = cb_f32[(size_t)idx[b] * D2 + c] - zf[o];
    p += d * d;
  }
  atomicAdd(&vqs, p);
  if (tid < BB) {
    int cnt = 0;
    for (int b2 = 0; b2 < BB; ++b2) cnt += (idx[b2] == idx[tid]);
    float pv = (float)cnt / (float)BB;
    atomicAdd(&ent, -(1.f / BB) * logf(pv + 1e-10f));
  }
  if (tid < BB * 3) {
    int b = tid / 3, g = tid % 3;
    const float* w = gate_w + g * D2;
    const float* z = zf + b * D2;
    float s = gate_b[g];
    for (int c = 0; c < D2; ++c) s += z[c] * w[c];
    gl[b][g] = s;
  }
  __syncthreads();
  if (tid < BB) {
    float a0 = gl[tid][0], a1 = gl[tid][1], a2 = gl[tid][2];
    float mx = fmaxf(a0, fmaxf(a1, a2));
    float e0 = expf(a0 - mx), e1 = expf(a1 - mx), e2 = expf(a2 - mx);
    float inv = 1.f / (e0 + e1 + e2);
    gl[tid][0] = e0 * inv; gl[tid][1] = e1 * inv; gl[tid][2] = e2 * inv;
  }
  __syncthreads();
  float vq_new = 1.25f * vqs / (float)(BB * D2);
  float halt   = 1.f / (1.f + expf(-(4.f - vq_new)));
  if (act) {
    for (int o = tid; o < BB * D2; o += 256) {
      int b = o >> 7, rem = o & 127, im = rem >> 6, d = rem & 63;
      float zq = cb_f32[(size_t)idx[b] * D2 + im * DD + d];
      float a  = im ? attn_i[b * DD + d] : attn_r[b * DD + d];
      float rd = im ? read_i[b * DD + d] : read_r[b * DD + d];
      float u  = gl[b][0] * a + gl[b][1] * rd + gl[b][2] * zq;
      float n  = im ? ni[b * DD + d] : nr[b * DD + d];
      float nv = 0.5f * n + 0.5f * u;
      if (im) gi[b * DD + d] = nv; else gr[b * DD + d] = nv;
    }
    if (tid < BB) {
      float c = cum[tid];
      pen[tid] += (1.f - c) * 0.01f;       // PONDER_P, pre-update cum
      cum[tid]  = c + (1.f - c) * halt;
    }
    if (tid == 0) { scal[3] = vq_new; scal[4] = expf(ent); }
  }
  __syncthreads();
  if (tid < BB) {
    atomicAdd(&cmean, cum[tid] * (1.f / BB));
    atomicAdd(&pensum, pen[tid]);
  }
  __syncthreads();
  if (tid == 0) {
    if (act && cmean > 0.95f) *activep = 0u;   // HALT_TH
    if (r == MAXREC - 1) {
      scal[0] += scal[3];                      // tvq  += vql
      scal[1] += pensum / (float)BB;           // tpond += mean(pen)
      scal[2]  = scal[4];                      // ppx (last timestep wins)
    }
  }
}

// ---------- end of timestep: memory slot write + ys ----------
__global__ void k_write(int t, const float* gr, const float* gi,
                        float* mem_r, float* mem_i, float* outb) {
  int i = blockIdx.x * 256 + threadIdx.x;
  if (i < BB * DD) {
    int b = i >> 6, d = i & 63;
    mem_r[((size_t)b * SLOTS + t) * DD + d] = gr[i];  // slot write == shift (perm-invariant read)
    mem_i[((size_t)b * SLOTS + t) * DD + d] = gi[i];
    outb[((size_t)b * TT + t) * D2 + d]      = gr[i];
    outb[((size_t)b * TT + t) * D2 + DD + d] = gi[i];
  }
}

// ---------- decoder GEMM: logits = out @ dec_w^T + dec_b (WMMA bf16) ----------
// A-tile (64x128 bf16) async-staged to LDS once per block, reused by 4 waves x 4 row tiles.
__global__ void k_dec(const __bf16* A, const __bf16* W, const float* bias, float* out) {
  __shared__ __attribute__((aligned(16))) __bf16 As[64 * D2];    // 16KB
  int tid = threadIdx.x;
  int wave = tid >> 5, lane = tid & 31, half = lane >> 4, l15 = lane & 15;
  int n0 = blockIdx.x * 64 + wave * 16;
  int mb = blockIdx.y * 64;
  stage_tile_16k(A + (size_t)mb * D2, (unsigned)(uintptr_t)&As[0], tid);
  const __bf16* Al = (const __bf16*)As;
  int n  = n0 + l15;
  float bn = bias[n];
  __builtin_prefetch(W + (size_t)n0 * D2, 0, 1);
#pragma unroll
  for (int mt = 0; mt < 4; ++mt) {
    v8f acc = {};
#pragma unroll
    for (int kk = 0; kk < 4; ++kk) {
      v16bf a = frag_a(Al, D2, mt * 16, kk * 32);   // ds_load from staged tile
      v16bf b = frag_b(W, D2, n0, kk * 32);
      acc = __builtin_amdgcn_wmma_f32_16x16x32_bf16(false, a, false, b,
                                                    (short)0, acc, false, false);
    }
#pragma unroll
    for (int r = 0; r < 8; ++r) {
      int m = mb + mt * 16 + r + 8 * half;
      out[(size_t)m * VV + n] = acc[r] + bn;
    }
  }
}

__global__ void k_scal(const float* scal, float* tail) {
  if (threadIdx.x == 0) { tail[0] = scal[0]; tail[1] = scal[1]; tail[2] = scal[2]; }
}

// ---------- host ----------
extern "C" void kernel_launch(void* const* d_in, const int* in_sizes, int n_in,
                              void* d_out, int out_size, void* d_ws, size_t ws_size,
                              hipStream_t stream) {
  const int*   x_seq   = (const int*)  d_in[0];
  const float* emb     = (const float*)d_in[1];
  const float* cb_f32  = (const float*)d_in[2];
  const float* qkv_wr  = (const float*)d_in[3];
  const float* qkv_br  = (const float*)d_in[4];
  const float* qkv_wi  = (const float*)d_in[5];
  const float* qkv_bi  = (const float*)d_in[6];
  const float* nrw     = (const float*)d_in[7];
  const float* nrb     = (const float*)d_in[8];
  const float* niw     = (const float*)d_in[9];
  const float* nib     = (const float*)d_in[10];
  const float* gate_w  = (const float*)d_in[11];
  const float* gate_b  = (const float*)d_in[12];
  const float* dec_w   = (const float*)d_in[13];
  const float* dec_b   = (const float*)d_in[14];
  float* logits = (float*)d_out;

  char* p = (char*)d_ws;
  auto alloc = [&](size_t bytes) -> void* {
    void* r = (void*)p;
    p += (bytes + 255) & ~(size_t)255;
    return r;
  };
  __bf16* cb_bf  = (__bf16*)alloc((size_t)VV * D2 * 2);
  __bf16* dw_bf  = (__bf16*)alloc((size_t)VV * D2 * 2);
  float*  c2     = (float*) alloc((size_t)VV * 4);
  __bf16* wrb    = (__bf16*)alloc(3 * DD * DD * 2);
  __bf16* wib    = (__bf16*)alloc(3 * DD * DD * 2);
  __bf16* nwib   = (__bf16*)alloc(3 * DD * DD * 2);
  float*  gr     = (float*) alloc(BB * DD * 4);
  float*  gi     = (float*) alloc(BB * DD * 4);
  float*  nr     = (float*) alloc(BB * DD * 4);
  float*  ni     = (float*) alloc(BB * DD * 4);
  __bf16* nr_bf  = (__bf16*)alloc(BB * DD * 2);
  __bf16* ni_bf  = (__bf16*)alloc(BB * DD * 2);
  float*  zf     = (float*) alloc(BB * D2 * 4);
  __bf16* zfb    = (__bf16*)alloc(BB * D2 * 2);
  float*  zf2    = (float*) alloc(BB * 4);
  float*  qkvbuf = (float*) alloc(6 * BB * DD * 4);
  float*  attn_r = (float*) alloc(BB * DD * 4);
  float*  attn_i = (float*) alloc(BB * DD * 4);
  float*  read_r = (float*) alloc(BB * DD * 4);
  float*  read_i = (float*) alloc(BB * DD * 4);
  unsigned long long* keys = (unsigned long long*)alloc(BB * 8);
  float*  mem_r  = (float*) alloc((size_t)BB * SLOTS * DD * 4);
  float*  mem_i  = (float*) alloc((size_t)BB * SLOTS * DD * 4);
  float*  outb   = (float*) alloc((size_t)BT * D2 * 4);
  __bf16* out_bf = (__bf16*)alloc((size_t)BT * D2 * 2);
  float*  cum    = (float*) alloc(BB * 4);
  float*  pen    = (float*) alloc(BB * 4);
  float*  scal   = (float*) alloc(8 * 4);       // tvq, tpond, ppx, vql, ppx_t
  unsigned int* activep = (unsigned int*)alloc(4);

  int ncb = VV * D2;
  k_cvt_bf16<<<(ncb + 255) / 256, 256, 0, stream>>>(cb_f32, cb_bf, ncb);
  k_cvt_bf16<<<(ncb + 255) / 256, 256, 0, stream>>>(dec_w, dw_bf, ncb);
  k_c2<<<(VV + 255) / 256, 256, 0, stream>>>(cb_f32, c2);
  int nw = 3 * DD * DD;
  k_cvt_bf16<<<(nw + 255) / 256, 256, 0, stream>>>(qkv_wr, wrb, nw);
  k_cvt_bf16<<<(nw + 255) / 256, 256, 0, stream>>>(qkv_wi, wib, nw);
  k_cvt_neg_bf16<<<(nw + 255) / 256, 256, 0, stream>>>(qkv_wi, nwib, nw);
  k_zero<<<(BB * DD + 255) / 256, 256, 0, stream>>>(gr, BB * DD);
  k_zero<<<(BB * DD + 255) / 256, 256, 0, stream>>>(gi, BB * DD);
  int nmem = BB * SLOTS * DD;
  k_zero<<<(nmem + 255) / 256, 256, 0, stream>>>(mem_r, nmem);
  k_zero<<<(nmem + 255) / 256, 256, 0, stream>>>(mem_i, nmem);
  k_zero<<<1, 256, 0, stream>>>(scal, 8);

  for (int t = 0; t < TT; ++t) {
    k_outer<<<16, 256, 0, stream>>>(t, x_seq, emb, gr, gi, cum, pen, scal, activep);
    for (int r = 0; r < MAXREC; ++r) {
      k_pre<<<1, 128, 0, stream>>>(gr, gi, nrw, nrb, niw, nib,
                                   nr, ni, nr_bf, ni_bf, zf, zfb, zf2, keys);
      k_qkv<<<1, 256, 0, stream>>>(nr_bf, ni_bf, wrb, wib, nwib,
                                   qkv_br, qkv_bi, qkvbuf);
      k_mid<<<BB, 256, 0, stream>>>(nr, ni, qkvbuf, mem_r, mem_i,
                                    attn_r, attn_i, read_r, read_i);
      k_vq<<<VV / 64, 128, 0, stream>>>(zfb, zf2, cb_bf, c2, keys);
      k_post<<<1, 256, 0, stream>>>(r, zf, cb_f32, keys, gate_w, gate_b,
                                    attn_r, attn_i, read_r, read_i,
                                    nr, ni, gr, gi, cum, pen, scal, activep);
    }
    k_write<<<16, 256, 0, stream>>>(t, gr, gi, mem_r, mem_i, outb);
  }

  int nout = BT * D2;
  k_cvt_bf16<<<(nout + 255) / 256, 256, 0, stream>>>(outb, out_bf, nout);
  k_dec<<<dim3(VV / 64, BT / 64), 128, 0, stream>>>(out_bf, dw_bf, dec_b, logits);
  k_scal<<<1, 32, 0, stream>>>(scal, logits + LOGITS_N);
}
